// GlmAsrAttention_14078902796647
// MI455X (gfx1250) — compile-verified, matrix operand order checked by reading, not run on hardware
//
#include <hip/hip_runtime.h>
#include <cstddef>

// ---------------- problem constants (match reference) ----------------
#define BB   2
#define SS   2048
#define DD   1024
#define HH   16
#define KVH  4
#define HDIM 64
#define GQ   (HH / KVH)       // 4
#define ROT  32

#define LPITCH 40             // LDS row pitch in ushorts (80B, 16B aligned)

typedef __attribute__((ext_vector_type(16))) __bf16 bf16x16;
typedef __attribute__((ext_vector_type(8)))  float  f32x8;

union ATile { bf16x16 v; uint4 q[2]; };

// fp32 -> bf16 RNE, scalar (returns raw bits)
static __device__ __forceinline__ unsigned short bf16bits(float f) {
    unsigned int u = __builtin_bit_cast(unsigned int, f);
    u += 0x7FFFu + ((u >> 16) & 1u);
    return (unsigned short)(u >> 16);
}

// pack two fp32 -> {hi:bf16(b), lo:bf16(a)} in one dword via v_perm_b32
static __device__ __forceinline__ unsigned int pack2bf(float a, float b) {
    unsigned int ua = __builtin_bit_cast(unsigned int, a);
    unsigned int ub = __builtin_bit_cast(unsigned int, b);
    ua += 0x7FFFu + ((ua >> 16) & 1u);
    ub += 0x7FFFu + ((ub >> 16) & 1u);
    return __builtin_amdgcn_perm(ub, ua, 0x07060302u);
}

static __device__ __forceinline__ f32x8 wmma_bf16(bf16x16 a, bf16x16 b, f32x8 c) {
    return __builtin_amdgcn_wmma_f32_16x16x32_bf16(
        false, a, false, b, (short)0, c, false, false);
}

// Read one 16x32 bf16 WMMA operand tile from an LDS slab (row-major,
// LPITCH-ushort pitch). lane r = row, hh = lane>>4 selects K-half:
// elements 0..7 <- K = hh*8..+7, elements 8..15 <- K = 16+hh*8..+7.
static __device__ __forceinline__ bf16x16
lds_tile(const unsigned short* slab, int row0, int r, int hh) {
    const unsigned short* p = slab + (size_t)(row0 + r) * LPITCH + hh * 8;
    ATile t;
    t.q[0] = *(const uint4*)(p);
    t.q[1] = *(const uint4*)(p + 16);
    return t.v;
}

// Generic batched-head GEMM: C[z] = alpha * A[z] * op(B[z]) + bias
// z -> (batch = z / Hpb, head = z % Hpb); B uses head / bHeadDiv (GQA).
// Block = 128 threads = 4 waves computing a 64x64 C macro-tile (2x2 of 32x32).
// Software-pipelined: global fetch of K-slab k+32 overlaps LDS reads + WMMA
// of slab k; fp32->bf16 conversion happens once per block at LDS-store time.
template <int TRANSB>
__global__ void __launch_bounds__(128)
gemm_wmma_bf16(
    const float* __restrict__ A, int lda, long long aBs, long long aHs,
    const float* __restrict__ Bm, int ldb, long long bBs, long long bHs, int bHeadDiv,
    const float* __restrict__ bias, float alpha,
    float* __restrict__ C, int ldc, long long cBs, long long cHs,
    int M, int N, int Kd, int Hpb) {

    __shared__ __align__(16) unsigned short Abf[64 * LPITCH];
    __shared__ __align__(16) unsigned short Bbf[64 * LPITCH];

    int z = blockIdx.z;
    int b = z / Hpb;
    int h = z % Hpb;
    const float* Ab = A + (size_t)b * aBs + (size_t)h * aHs;
    const float* Bb = Bm + (size_t)b * bBs + (size_t)(h / bHeadDiv) * bHs;
    float*       Cb = C + (size_t)b * cBs + (size_t)h * cHs;

    int lane = threadIdx.x;               // 0..31 (wave32)
    int tid  = threadIdx.y * 32 + lane;   // 0..127
    int r    = lane & 15;
    int hh   = lane >> 4;
    int wm   = threadIdx.y & 1;           // wave's 32x32 subtile in 64x64
    int wn   = threadIdx.y >> 1;
    int tileMB = blockIdx.y * 64;
    int tileNB = blockIdx.x * 64;

    // K-invariant addressing for the cooperative slab fill (4 float4 each
    // for A and B per thread; 512 float4 per 64x32 slab).
    int aRow[4], aC4[4];                  // A: row-major 64 x 32
    const float* aPtr[4];
    unsigned short* aSt[4];
#pragma unroll
    for (int t = 0; t < 4; ++t) {
        int idx = tid + t * 128;
        aRow[t] = idx >> 3;
        aC4[t]  = idx & 7;
        aPtr[t] = Ab + (size_t)(tileMB + aRow[t]) * lda + aC4[t] * 4;
        aSt[t]  = &Abf[aRow[t] * LPITCH + aC4[t] * 4];
    }
    const float* bPtr[4];
    unsigned short* bSt[4];
    int bK[4];                            // TRANSB==0 transpose-store K row
#pragma unroll
    for (int t = 0; t < 4; ++t) {
        int idx = tid + t * 128;
        if (TRANSB) {                     // B given as [N][K]: same as A
            int row = idx >> 3, c4 = idx & 7;
            bPtr[t] = Bb + (size_t)(tileNB + row) * ldb + c4 * 4;
            bSt[t]  = &Bbf[row * LPITCH + c4 * 4];
            bK[t]   = 0;
        } else {                          // B given as [K][N]: 32 K x 64 N
            int krow = idx >> 4, n4 = idx & 15;
            bPtr[t] = Bb + (size_t)krow * ldb + tileNB + n4 * 4;
            bSt[t]  = &Bbf[(n4 * 4) * LPITCH + krow];
            bK[t]   = krow;
        }
    }

    f32x8 c00 = {}, c01 = {}, c10 = {}, c11 = {};
    float4 fa[4], fb[4];                  // staged next-slab registers

    // prologue fetch: k0 = 0 (loads issue back-to-back, no interleaved waits)
#pragma unroll
    for (int t = 0; t < 4; ++t) fa[t] = *(const float4*)(aPtr[t]);
#pragma unroll
    for (int t = 0; t < 4; ++t)
        fb[t] = *(const float4*)(bPtr[t] + (TRANSB ? 0 : 0));

    for (int k0 = 0; k0 < Kd; k0 += 32) {
        __syncthreads();                  // previous compute done; slab reusable
        // ---- convert staged regs -> bf16 LDS slabs ----
#pragma unroll
        for (int t = 0; t < 4; ++t) {
            uint2 pv;
            pv.x = pack2bf(fa[t].x, fa[t].y);
            pv.y = pack2bf(fa[t].z, fa[t].w);
            *(uint2*)aSt[t] = pv;
        }
        if (TRANSB) {
#pragma unroll
            for (int t = 0; t < 4; ++t) {
                uint2 pv;
                pv.x = pack2bf(fb[t].x, fb[t].y);
                pv.y = pack2bf(fb[t].z, fb[t].w);
                *(uint2*)bSt[t] = pv;
            }
        } else {
#pragma unroll
            for (int t = 0; t < 4; ++t) {
                float ff[4] = { fb[t].x, fb[t].y, fb[t].z, fb[t].w };
#pragma unroll
                for (int j = 0; j < 4; ++j)
                    bSt[t][j * LPITCH] = bf16bits(ff[j]);
            }
        }
        __syncthreads();                  // slab ready for all 4 waves

        // ---- fetch NEXT slab; loads stay in flight across the WMMAs ----
        if (k0 + 32 < Kd) {
            int kn = k0 + 32;
#pragma unroll
            for (int t = 0; t < 4; ++t)
                fa[t] = *(const float4*)(aPtr[t] + kn);
#pragma unroll
            for (int t = 0; t < 4; ++t)
                fb[t] = *(const float4*)(bPtr[t] + (TRANSB ? kn : (size_t)kn * ldb));
        }

        // ---- compute: 8x ds_load_b128 + 4x WMMA per K-step per wave ----
        bf16x16 a0 = lds_tile(Abf, wm * 32,      r, hh);
        bf16x16 a1 = lds_tile(Abf, wm * 32 + 16, r, hh);
        bf16x16 b0 = lds_tile(Bbf, wn * 32,      r, hh);
        bf16x16 b1 = lds_tile(Bbf, wn * 32 + 16, r, hh);
        c00 = wmma_bf16(a0, b0, c00);
        c01 = wmma_bf16(a0, b1, c01);
        c10 = wmma_bf16(a1, b0, c10);
        c11 = wmma_bf16(a1, b1, c11);
    }

    int tileM = tileMB + wm * 32;
    int tileN = tileNB + wn * 32;
    float bv0 = bias ? bias[tileN + r]      : 0.0f;
    float bv1 = bias ? bias[tileN + 16 + r] : 0.0f;
#pragma unroll
    for (int v = 0; v < 8; ++v) {
        int m0 = tileM + v + 8 * hh;   // C layout: VGPR v -> M = v + 8*laneHalf
        Cb[(size_t)m0        * ldc + tileN + r]      = alpha * c00[v] + bv0;
        Cb[(size_t)m0        * ldc + tileN + 16 + r] = alpha * c01[v] + bv1;
        Cb[(size_t)(m0 + 16) * ldc + tileN + r]      = alpha * c10[v] + bv0;
        Cb[(size_t)(m0 + 16) * ldc + tileN + 16 + r] = alpha * c11[v] + bv1;
    }
}

// Partial RoPE on first ROT=32 dims of each Q head and K head, in place.
__global__ void rope_kernel(float* __restrict__ qbuf, float* __restrict__ kbuf,
                            const float* __restrict__ cosp,
                            const float* __restrict__ sinp) {
    const int qElems = BB * SS * HH * 16;            // one thread per dim-pair
    const int total  = qElems + BB * SS * KVH * 16;
    int idx = blockIdx.x * blockDim.x + threadIdx.x;
    if (idx >= total) return;

    float* base;
    int t, d;
    if (idx < qElems) {
        d = idx & 15;
        int h = (idx >> 4) & (HH - 1);
        t = idx >> 8;                                // token = b*S + s
        base = qbuf + (size_t)t * (HH * HDIM) + h * HDIM;
    } else {
        int j = idx - qElems;
        d = j & 15;
        int kv = (j >> 4) & (KVH - 1);
        t = j >> 6;
        base = kbuf + (size_t)t * (KVH * HDIM) + kv * HDIM;
    }
    const float* c = cosp + (size_t)t * ROT;
    const float* s = sinp + (size_t)t * ROT;
    float x1 = base[d], x2 = base[d + 16];
    base[d]      = x1 * c[d]      - x2 * s[d];
    base[d + 16] = x2 * c[d + 16] + x1 * s[d + 16];
}

// In-place fp32 softmax over rows of length 2048; one block per row.
__global__ void softmax_rows_2048(float* __restrict__ data) {
    float* p = data + (size_t)blockIdx.x * SS;
    int tid = threadIdx.x;  // 256 threads, 8 elems each
    float vals[8];
    float m = -3.402823466e38f;
#pragma unroll
    for (int i = 0; i < 8; ++i) {
        vals[i] = p[tid + i * 256];
        m = fmaxf(m, vals[i]);
    }
    __shared__ float red[256];
    red[tid] = m; __syncthreads();
    for (int s = 128; s > 0; s >>= 1) {
        if (tid < s) red[tid] = fmaxf(red[tid], red[tid + s]);
        __syncthreads();
    }
    m = red[0]; __syncthreads();
    float sum = 0.0f;
#pragma unroll
    for (int i = 0; i < 8; ++i) { vals[i] = __expf(vals[i] - m); sum += vals[i]; }
    red[tid] = sum; __syncthreads();
    for (int s = 128; s > 0; s >>= 1) {
        if (tid < s) red[tid] += red[tid + s];
        __syncthreads();
    }
    float inv = 1.0f / red[0];
#pragma unroll
    for (int i = 0; i < 8; ++i) p[tid + i * 256] = vals[i] * inv;
}

extern "C" void kernel_launch(void* const* d_in, const int* in_sizes, int n_in,
                              void* d_out, int out_size, void* d_ws, size_t ws_size,
                              hipStream_t stream) {
    (void)in_sizes; (void)n_in; (void)out_size; (void)ws_size;
    const float* x    = (const float*)d_in[0];
    const float* cosp = (const float*)d_in[1];
    const float* sinp = (const float*)d_in[2];
    const float* q_w  = (const float*)d_in[3];
    const float* q_b  = (const float*)d_in[4];
    const float* k_w  = (const float*)d_in[5];
    const float* v_w  = (const float*)d_in[6];
    const float* v_b  = (const float*)d_in[7];
    const float* o_w  = (const float*)d_in[8];
    const float* o_b  = (const float*)d_in[9];

    float* attn_out = (float*)d_out;                              // B*S*D
    float* attn_w   = attn_out + (size_t)BB * SS * DD;            // B*H*S*S

    const int T = BB * SS;                                        // 4096 tokens
    float* qbuf = (float*)d_ws;                                   // T x 1024
    float* kbuf = qbuf + (size_t)T * (HH * HDIM);                 // T x 256
    float* vbuf = kbuf + (size_t)T * (KVH * HDIM);                // T x 256
    float* ctx  = qbuf;  // reuse q after scores consume it       // T x 1024

    dim3 blk(32, 4);  // 4 waves = one 64x64 C macro-tile per block

    // --- QKV projections: [4096 x 1024] x weights -----------------------
    gemm_wmma_bf16<0><<<dim3((HH*HDIM)/64, T/64, 1), blk, 0, stream>>>(
        x, DD, 0, 0,  q_w, HH*HDIM, 0, 0, 1,  q_b, 1.0f,
        qbuf, HH*HDIM, 0, 0,  T, HH*HDIM, DD, 1);
    gemm_wmma_bf16<0><<<dim3((KVH*HDIM)/64, T/64, 1), blk, 0, stream>>>(
        x, DD, 0, 0,  k_w, KVH*HDIM, 0, 0, 1,  nullptr, 1.0f,
        kbuf, KVH*HDIM, 0, 0,  T, KVH*HDIM, DD, 1);
    gemm_wmma_bf16<0><<<dim3((KVH*HDIM)/64, T/64, 1), blk, 0, stream>>>(
        x, DD, 0, 0,  v_w, KVH*HDIM, 0, 0, 1,  v_b, 1.0f,
        vbuf, KVH*HDIM, 0, 0,  T, KVH*HDIM, DD, 1);

    // --- RoPE on Q and K -------------------------------------------------
    {
        int total = BB * SS * (HH + KVH) * 16;
        rope_kernel<<<(total + 255) / 256, 256, 0, stream>>>(qbuf, kbuf, cosp, sinp);
    }

    // --- scores = scale * Q @ K^T  -> attn_weights region ---------------
    gemm_wmma_bf16<1><<<dim3(SS/64, SS/64, BB*HH), blk, 0, stream>>>(
        qbuf, HH*HDIM, (long long)SS*HH*HDIM, (long long)HDIM,
        kbuf, KVH*HDIM, (long long)SS*KVH*HDIM, (long long)HDIM, GQ,
        nullptr, 0.125f /* 1/sqrt(64) */,
        attn_w, SS, (long long)HH*SS*SS, (long long)SS*SS,
        SS, SS, HDIM, HH);

    // --- softmax in place on attn_weights --------------------------------
    softmax_rows_2048<<<BB * HH * SS, 256, 0, stream>>>(attn_w);

    // --- context = probs @ V  (written as (B,S,H,HD) into ctx) -----------
    gemm_wmma_bf16<0><<<dim3(HDIM/64, SS/64, BB*HH), blk, 0, stream>>>(
        attn_w, SS, (long long)HH*SS*SS, (long long)SS*SS,
        vbuf, KVH*HDIM, (long long)SS*KVH*HDIM, (long long)HDIM, GQ,
        nullptr, 1.0f,
        ctx, HH*HDIM, (long long)SS*HH*HDIM, (long long)HDIM,
        SS, HDIM, SS, HH);

    // --- output projection: ctx @ o_w + o_b ------------------------------
    gemm_wmma_bf16<0><<<dim3(DD/64, T/64, 1), blk, 0, stream>>>(
        ctx, HH*HDIM, 0, 0,  o_w, DD, 0, 0, 1,  o_b, 1.0f,
        attn_out, DD, 0, 0,  T, DD, HH*HDIM, 1);
}
